// ResidualGumbelVQ_65953517797734
// MI455X (gfx1250) — compile-verified, matrix operand order checked by reading, not run on hardware
//
#include <hip/hip_runtime.h>
#include <hip/hip_bf16.h>
#include <math.h>

// Problem constants (from reference): N=65536, K=4096, D=512
#define NROWS 65536
#define KCOLS 4096
#define DDIM  512

typedef __attribute__((ext_vector_type(16))) __bf16 v16bf;
typedef __attribute__((ext_vector_type(8)))  __bf16 v8bf;
typedef __attribute__((ext_vector_type(8)))  float  v8f;

// ---- workspace layout (bytes) --------------------------------------------
#define WS_EMB_NB   0                              // K*D bf16 normalized emb (4 MB)
#define WS_INV_ZE   (KCOLS * DDIM * 2)             // N f32 inverse row norms of z_e
#define WS_ROW_M    (WS_INV_ZE + NROWS * 4)        // N f32 row max (logits)
#define WS_ROW_S    (WS_ROW_M + NROWS * 4)         // N f32 row sum-exp
#define WS_ROW_IDX  (WS_ROW_S + NROWS * 4)         // N i32 argmax
#define WS_AVG      (WS_ROW_IDX + NROWS * 4)       // K f32 avg_probs accumulator

// ---- kernel 1: L2-normalize embeddings -> bf16 (B matrix, L2-resident) ---
__global__ void vq_normalize_emb(const float* __restrict__ emb,
                                 __bf16* __restrict__ emb_nb) {
  __shared__ float sred[256];
  const int k = blockIdx.x;
  const int t = threadIdx.x;
  const float* row = emb + (size_t)k * DDIM;
  float e0 = row[t], e1 = row[t + 256];
  sred[t] = e0 * e0 + e1 * e1;
  __syncthreads();
  for (int s = 128; s > 0; s >>= 1) {
    if (t < s) sred[t] += sred[t + s];
    __syncthreads();
  }
  const float inv = 1.0f / fmaxf(sqrtf(sred[0]), 1e-12f);
  __bf16* orow = emb_nb + (size_t)k * DDIM;
  orow[t]       = (__bf16)(e0 * inv);
  orow[t + 256] = (__bf16)(e1 * inv);
}

// ---- kernel 2: inverse L2 norm per z_e row (one wave32 per row) ----------
__global__ void vq_rownorm_ze(const float* __restrict__ z_e,
                              float* __restrict__ inv_ze) {
  const int t = threadIdx.x;
  const int wave = t >> 5, lane = t & 31;
  const int n = blockIdx.x * 8 + wave;
  const float* row = z_e + (size_t)n * DDIM;
  float ss = 0.0f;
#pragma unroll
  for (int i = 0; i < 16; ++i) { float v = row[lane + i * 32]; ss += v * v; }
#pragma unroll
  for (int m = 16; m >= 1; m >>= 1) ss += __shfl_xor(ss, m, 32);
  if (lane == 0) inv_ze[n] = 1.0f / fmaxf(sqrtf(ss), 1e-12f);
}

__global__ void vq_zero_avg(float* __restrict__ avg_acc) {
  avg_acc[blockIdx.x * 256 + threadIdx.x] = 0.0f;
}

// ---- main fused GEMM: 128 rows/block, 8 waves x 16 rows, all K columns ---
// PASS 0: online (max, sum-exp, argmax) per row -> row_m/row_s/row_idx
// PASS 1: recompute logits, accumulate avg_probs[k] += exp(l-m)/s
// B tile double-buffered in LDS; next tile's global loads issued before the
// WMMA loop so L2 latency hides behind 16 chained v_wmma ops; one barrier
// per iteration.
template <int PASS>
__global__ __launch_bounds__(256) void vq_logits_pass(
    const float* __restrict__ z_e, const __bf16* __restrict__ emb_nb,
    const float* __restrict__ inv_ze, const float* __restrict__ logit_scale,
    float* __restrict__ row_m, float* __restrict__ row_s,
    int* __restrict__ row_idx, float* __restrict__ avg_acc) {
  __shared__ __bf16 Bs[2][16 * DDIM];  // 2 x 16 KB double-buffered B tile
  __shared__ float  avg_l[KCOLS];      // 16 KB per-block avg_probs partial

  const int t = threadIdx.x;
  const int wave = t >> 5, lane = t & 31;
  const int hi = lane >> 4;            // 0: lanes 0-15, 1: lanes 16-31
  const int cl = lane & 15;            // A-row-in-tile / C-column-in-tile
  const int m0 = blockIdx.x * 128 + wave * 16;
  const int r  = m0 + cl;
  const float scale = logit_scale[0];

  if (PASS == 1) {
    for (int i = t; i < KCOLS; i += 256) avg_l[i] = 0.0f;
  }

  // Build A fragments once; keep resident in VGPRs (16 x v16bf = 128 VGPRs).
  // ISA 7.12.2 16-bit A 16x32: lanes 0-15 k={0..7,16..23}, lanes 16-31 +8.
  const float* zrow = z_e + (size_t)r * DDIM;
  const float inv = inv_ze[r];
  v16bf afrag[16];
#pragma unroll
  for (int kt = 0; kt < 16; ++kt) {
    const int base = kt * 32 + hi * 8;
    float4 f0 = *(const float4*)(zrow + base);
    float4 f1 = *(const float4*)(zrow + base + 4);
    float4 f2 = *(const float4*)(zrow + base + 16);
    float4 f3 = *(const float4*)(zrow + base + 20);
    v16bf a;
    a[0]  = (__bf16)(f0.x * inv); a[1]  = (__bf16)(f0.y * inv);
    a[2]  = (__bf16)(f0.z * inv); a[3]  = (__bf16)(f0.w * inv);
    a[4]  = (__bf16)(f1.x * inv); a[5]  = (__bf16)(f1.y * inv);
    a[6]  = (__bf16)(f1.z * inv); a[7]  = (__bf16)(f1.w * inv);
    a[8]  = (__bf16)(f2.x * inv); a[9]  = (__bf16)(f2.y * inv);
    a[10] = (__bf16)(f2.z * inv); a[11] = (__bf16)(f2.w * inv);
    a[12] = (__bf16)(f3.x * inv); a[13] = (__bf16)(f3.y * inv);
    a[14] = (__bf16)(f3.z * inv); a[15] = (__bf16)(f3.w * inv);
    afrag[kt] = a;
  }

  // C/D layout: lane owns column cl, VGPR j owns row m0 + j + hi*8.
  float mx[8], sm[8], rm[8], rsi[8];
  int   ix[8];
#pragma unroll
  for (int j = 0; j < 8; ++j) { mx[j] = -__builtin_inff(); sm[j] = 0.0f; ix[j] = 0; }
  if (PASS == 1) {
#pragma unroll
    for (int j = 0; j < 8; ++j) {
      const int rr = m0 + j + hi * 8;
      rm[j]  = row_m[rr];
      rsi[j] = 1.0f / row_s[rr];
    }
  }

  // ---- prologue: stage tile 0 into buffer 0 ----
  uint4 pref[4];
  {
    const uint4* src = (const uint4*)(emb_nb);
#pragma unroll
    for (int i = 0; i < 4; ++i) pref[i] = src[t * 4 + i];
    uint4* dst = (uint4*)(&Bs[0][0]);
#pragma unroll
    for (int i = 0; i < 4; ++i) dst[t * 4 + i] = pref[i];
  }
  __syncthreads();  // also covers avg_l zeroing for PASS==1

  for (int nb = 0; nb < KCOLS / 16; ++nb) {
    const int cur = nb & 1;
    const int nxt = cur ^ 1;
    const int nb_next = (nb + 1 < KCOLS / 16) ? nb + 1 : (KCOLS / 16 - 1);

    // issue next tile's global loads now; consumed only at the ds_store below
    {
      const uint4* src =
          (const uint4*)(emb_nb + (size_t)nb_next * (16 * DDIM));
#pragma unroll
      for (int i = 0; i < 4; ++i) pref[i] = src[t * 4 + i];
    }

    // 16 chained WMMAs over D=512, split into two independent chains
    const __bf16* bbase = &Bs[cur][cl * DDIM + hi * 8];
    v8f acc0 = {}, acc1 = {};
#pragma unroll
    for (int kt = 0; kt < 16; kt += 2) {
      {
        v8bf blo = *(const v8bf*)(bbase + kt * 32);
        v8bf bhi = *(const v8bf*)(bbase + kt * 32 + 16);
        v16bf b = __builtin_shufflevector(blo, bhi,
                    0, 1, 2, 3, 4, 5, 6, 7, 8, 9, 10, 11, 12, 13, 14, 15);
        acc0 = __builtin_amdgcn_wmma_f32_16x16x32_bf16(
            false, afrag[kt], false, b, (short)0, acc0, false, false);
      }
      {
        v8bf blo = *(const v8bf*)(bbase + (kt + 1) * 32);
        v8bf bhi = *(const v8bf*)(bbase + (kt + 1) * 32 + 16);
        v16bf b = __builtin_shufflevector(blo, bhi,
                    0, 1, 2, 3, 4, 5, 6, 7, 8, 9, 10, 11, 12, 13, 14, 15);
        acc1 = __builtin_amdgcn_wmma_f32_16x16x32_bf16(
            false, afrag[kt + 1], false, b, (short)0, acc1, false, false);
      }
    }

    // store prefetched tile into the other buffer (load-wait lands here,
    // hidden behind the WMMA chain above)
    {
      uint4* dst = (uint4*)(&Bs[nxt][0]);
#pragma unroll
      for (int i = 0; i < 4; ++i) dst[t * 4 + i] = pref[i];
    }

    const int col = nb * 16 + cl;
    if (PASS == 0) {
#pragma unroll
      for (int j = 0; j < 8; ++j) {
        const float l = (acc0[j] + acc1[j]) * scale;
        // one-exp online softmax update
        const float e = __expf(-fabsf(l - mx[j]));
        const bool  g = l > mx[j];
        sm[j] = g ? (sm[j] * e + 1.0f) : (sm[j] + e);
        if (g) ix[j] = col;
        mx[j] = fmaxf(mx[j], l);
      }
    } else {
      float psum = 0.0f;
#pragma unroll
      for (int j = 0; j < 8; ++j)
        psum += __expf((acc0[j] + acc1[j]) * scale - rm[j]) * rsi[j];
      psum += __shfl_xor(psum, 16, 32);  // fold rows 8..15 onto lanes 0-15
      if (hi == 0) atomicAdd(&avg_l[col], psum);
    }

    __syncthreads();  // buffer nxt fully staged; everyone done with cur
  }

  if (PASS == 0) {
    // merge online-softmax state across the 16 lanes of each half-wave
#pragma unroll
    for (int j = 0; j < 8; ++j) {
#pragma unroll
      for (int msk = 1; msk <= 8; msk <<= 1) {
        const float om = __shfl_xor(mx[j], msk, 32);
        const float os = __shfl_xor(sm[j], msk, 32);
        const int   oi = __shfl_xor(ix[j], msk, 32);
        const float e  = __expf(-fabsf(om - mx[j]));
        const bool  g  = om > mx[j];
        sm[j] = g ? (sm[j] * e + os) : (sm[j] + os * e);
        if (g || (om == mx[j] && oi < ix[j])) ix[j] = oi;
        mx[j] = fmaxf(mx[j], om);
      }
    }
    if (cl == 0) {
#pragma unroll
      for (int j = 0; j < 8; ++j) {
        const int rr = m0 + j + hi * 8;
        row_m[rr]   = mx[j];
        row_s[rr]   = sm[j];
        row_idx[rr] = ix[j];
      }
    }
  } else {
    for (int i = t; i < KCOLS; i += 256) atomicAdd(&avg_acc[i], avg_l[i]);
  }
}

// ---- finalize scalars: perplexity + alpha --------------------------------
__global__ void vq_finalize(const float* __restrict__ avg_acc,
                            const float* __restrict__ residual_weight,
                            float* __restrict__ out_scalars) {
  __shared__ float sred[256];
  const int t = threadIdx.x;
  float p = 0.0f;
  for (int i = t; i < KCOLS; i += 256) {
    const float a = avg_acc[i] * (1.0f / (float)NROWS);
    p += a * logf(a + 1e-10f);
  }
  sred[t] = p;
  __syncthreads();
  for (int s = 128; s > 0; s >>= 1) {
    if (t < s) sred[t] += sred[t + s];
    __syncthreads();
  }
  if (t == 0) {
    out_scalars[0] = __expf(-sred[0]);                             // perplexity
    out_scalars[1] = 1.0f / (1.0f + __expf(-residual_weight[0]));  // alpha
  }
}

// ---- gather + residual blend (pure streaming, float4) --------------------
__global__ void vq_gather_blend(const float* __restrict__ z_e,
                                const float* __restrict__ emb,
                                const int* __restrict__ row_idx,
                                const float* __restrict__ residual_weight,
                                float* __restrict__ zq_out,
                                float* __restrict__ idx_out) {
  const int n = blockIdx.x;
  const int t = threadIdx.x;  // 128 threads x float4 = 512 floats
  const int idx = row_idx[n];
  const float alpha = 1.0f / (1.0f + __expf(-residual_weight[0]));
  const float4 ze4 = ((const float4*)(z_e + (size_t)n * DDIM))[t];
  const float4 eb4 = ((const float4*)(emb + (size_t)idx * DDIM))[t];
  float4 o;
  o.x = alpha * eb4.x + (1.0f - alpha) * ze4.x;
  o.y = alpha * eb4.y + (1.0f - alpha) * ze4.y;
  o.z = alpha * eb4.z + (1.0f - alpha) * ze4.z;
  o.w = alpha * eb4.w + (1.0f - alpha) * ze4.w;
  ((float4*)(zq_out + (size_t)n * DDIM))[t] = o;
  if (t == 0) idx_out[n] = (float)idx;
}

// ---- host launcher -------------------------------------------------------
extern "C" void kernel_launch(void* const* d_in, const int* in_sizes, int n_in,
                              void* d_out, int out_size, void* d_ws,
                              size_t ws_size, hipStream_t stream) {
  (void)in_sizes; (void)n_in; (void)out_size; (void)ws_size;
  const float* z_e             = (const float*)d_in[0];
  const float* emb             = (const float*)d_in[1];
  const float* logit_scale     = (const float*)d_in[2];
  const float* residual_weight = (const float*)d_in[3];
  float* out = (float*)d_out;

  char* ws = (char*)d_ws;
  __bf16* emb_nb  = (__bf16*)(ws + WS_EMB_NB);
  float*  inv_ze  = (float*)(ws + WS_INV_ZE);
  float*  row_m   = (float*)(ws + WS_ROW_M);
  float*  row_s   = (float*)(ws + WS_ROW_S);
  int*    row_idx = (int*)(ws + WS_ROW_IDX);
  float*  avg_acc = (float*)(ws + WS_AVG);

  float* zq_out      = out;
  float* idx_out     = out + (size_t)NROWS * DDIM;
  float* scalars_out = out + (size_t)NROWS * DDIM + NROWS;

  vq_normalize_emb<<<KCOLS, 256, 0, stream>>>(emb, emb_nb);
  vq_rownorm_ze<<<NROWS / 8, 256, 0, stream>>>(z_e, inv_ze);
  vq_zero_avg<<<KCOLS / 256, 256, 0, stream>>>(avg_acc);
  vq_logits_pass<0><<<NROWS / 128, 256, 0, stream>>>(
      z_e, emb_nb, inv_ze, logit_scale, row_m, row_s, row_idx, avg_acc);
  vq_logits_pass<1><<<NROWS / 128, 256, 0, stream>>>(
      z_e, emb_nb, inv_ze, logit_scale, row_m, row_s, row_idx, avg_acc);
  vq_finalize<<<1, 256, 0, stream>>>(avg_acc, residual_weight, scalars_out);
  vq_gather_blend<<<NROWS, 128, 0, stream>>>(z_e, emb, row_idx,
                                             residual_weight, zq_out, idx_out);
}
